// TopoEncoder_73993696575587
// MI455X (gfx1250) — compile-verified, matrix operand order checked by reading, not run on hardware
//
#include <hip/hip_runtime.h>

typedef _Float16 v16h __attribute__((ext_vector_type(16)));
typedef _Float16 v8h  __attribute__((ext_vector_type(8)));
typedef float    v8f  __attribute__((ext_vector_type(8)));

// Hard scheduling pin with DATA dependencies: forces all 8 fragment loads to
// complete before this point (asm reads them) and forces the WMMA chain to
// stay below (it consumes the asm outputs). Keeps all 8 fragments in distinct
// register tuples so the loads issue as one clause with 16 loads in flight.
__device__ __forceinline__ void pin8(v16h& b0, v16h& b1, v16h& b2, v16h& b3,
                                     v16h& b4, v16h& b5, v16h& b6, v16h& b7) {
  asm volatile("" : "+v"(b0), "+v"(b1), "+v"(b2), "+v"(b3),
                    "+v"(b4), "+v"(b5), "+v"(b6), "+v"(b7));
}

__device__ __forceinline__ v8f wmma_f16(v16h a, v16h b, v8f c) {
  // D = A(16x32 f16) x B(32x16 f16) + C(16x16 f32)
  return __builtin_amdgcn_wmma_f32_16x16x32_f16(false, a, false, b, (short)0, c,
                                                false, false);
}

// Build a 16x32 A/B fragment for this lane: two runs of 8 consecutive halves
// (k = kb..kb+7 and kb+16..kb+23). Works for LDS or global pointers.
__device__ __forceinline__ v16h load_frag(const _Float16* p) {
  v8h lo = *(const v8h*)p;
  v8h hi = *(const v8h*)(p + 16);
  v16h r;
#pragma unroll
  for (int i = 0; i < 8; ++i) { r[i] = lo[i]; r[i + 8] = hi[i]; }
  return r;
}

// B fragment from row-major weight W[N][K] (computing out = A @ W^T):
// B[k][n] = W[n][k]; lane holds column n, ks per the 16-bit B layout.
__device__ __forceinline__ v16h load_bfrag(const _Float16* W, int K, int n, int kb) {
  return load_frag(W + (size_t)n * K + kb);
}

// Prefetch an f16 array into the cache hierarchy (gfx1250 global_prefetch_b8).
__device__ __forceinline__ void prefetch_weights(const _Float16* W, int halves,
                                                 int tid, int nthreads) {
  for (int p = tid * 64; p < halves; p += nthreads * 64)
    __builtin_prefetch(W + p, 0, 3);
}

// ---------------------------------------------------------------- fp32 -> fp16
__global__ __launch_bounds__(256) void k_cvt(const float* __restrict__ s,
                                             _Float16* __restrict__ d, int n) {
  int i = blockIdx.x * 256 + threadIdx.x;
  if (i < n) d[i] = (_Float16)s[i];
}

// ---------------------------------------------------------------- WireNet RNN
// One wave handles 16 wires. h kept f32 in registers (C layout), mirrored to
// LDS as f16 each step to serve as the A operand of h @ W_hh^T.
__global__ __launch_bounds__(128) void k_rnn(
    const float* __restrict__ edges, const int* __restrict__ edge_index,
    const int* __restrict__ e_len, const float* __restrict__ h0,
    const _Float16* __restrict__ Wih, const _Float16* __restrict__ Whh,
    const float* __restrict__ b_ih, const float* __restrict__ b_hh,
    float* __restrict__ feat_wire, int n_w) {
  const int wv = threadIdx.x >> 5;
  const int lane = threadIdx.x & 31;
  const int ln = lane & 15, hi = lane >> 4;
  const int row0 = (blockIdx.x * 4 + wv) * 16;
  if (row0 >= n_w) return;

  // warm weight caches
  prefetch_weights(Wih, 128 * 128, threadIdx.x, 128);
  prefetch_weights(Whh, 128 * 128, threadIdx.x, 128);

  __shared__ _Float16 s_h[4][16][136];  // padded pitch: 136 halves
  _Float16(*H)[136] = s_h[wv];

  int lenm[8];
#pragma unroll
  for (int i = 0; i < 8; ++i) {
    int r = row0 + i + 8 * hi;
    lenm[i] = e_len[r < n_w ? r : n_w - 1];
  }

  float hreg[8][8];  // [ntile][elem] : C layout, row m = elem + 8*hi
#pragma unroll
  for (int nt = 0; nt < 8; ++nt)
#pragma unroll
    for (int i = 0; i < 8; ++i) {
      int r = row0 + i + 8 * hi;
      float v = h0[(size_t)(r < n_w ? r : n_w - 1) * 128 + nt * 16 + ln];
      hreg[nt][i] = v;
      H[i + 8 * hi][nt * 16 + ln] = (_Float16)v;
    }

  const int wireA = (row0 + ln) < n_w ? (row0 + ln) : n_w - 1;
  for (int t = 0; t < 16; ++t) {
    const int eidx = edge_index[wireA * 16 + t];
    const float* xrow = edges + (size_t)eidx * 128;
    v16h xa[4];  // x_t A fragments (from global, f32 -> f16)
#pragma unroll
    for (int kt = 0; kt < 4; ++kt) {
      const int kb = kt * 32 + 8 * hi;
      v16h a;
#pragma unroll
      for (int e = 0; e < 8; ++e) {
        a[e] = (_Float16)xrow[kb + e];
        a[e + 8] = (_Float16)xrow[kb + 16 + e];
      }
      xa[kt] = a;
    }
    v16h ha[4];  // h A fragments (from LDS mirror, previous step)
#pragma unroll
    for (int kt = 0; kt < 4; ++kt) ha[kt] = load_frag(&H[ln][kt * 32 + 8 * hi]);

    for (int nt = 0; nt < 8; ++nt) {
      const int n = nt * 16 + ln;
      // batch all 8 B fragments so the loads overlap, then run the WMMA chain
      v16h bf[8];
#pragma unroll
      for (int kt = 0; kt < 4; ++kt) {
        bf[kt] = load_bfrag(Wih, 128, n, kt * 32 + 8 * hi);
        bf[4 + kt] = load_bfrag(Whh, 128, n, kt * 32 + 8 * hi);
      }
      const float binit = b_ih[n] + b_hh[n];
      pin8(bf[0], bf[1], bf[2], bf[3], bf[4], bf[5], bf[6], bf[7]);
      v8f acc;
#pragma unroll
      for (int i = 0; i < 8; ++i) acc[i] = binit;
#pragma unroll
      for (int kt = 0; kt < 4; ++kt) {
        acc = wmma_f16(xa[kt], bf[kt], acc);
        acc = wmma_f16(ha[kt], bf[4 + kt], acc);
      }
#pragma unroll
      for (int i = 0; i < 8; ++i) {
        float nh = tanhf(acc[i]);
        if (t < lenm[i]) hreg[nt][i] = nh;
      }
    }
    if (t < 15) {
#pragma unroll
      for (int nt = 0; nt < 8; ++nt)
#pragma unroll
        for (int i = 0; i < 8; ++i)
          H[i + 8 * hi][nt * 16 + ln] = (_Float16)hreg[nt][i];
    }
  }
#pragma unroll
  for (int nt = 0; nt < 8; ++nt)
#pragma unroll
    for (int i = 0; i < 8; ++i) {
      int r = row0 + i + 8 * hi;
      if (r < n_w) feat_wire[(size_t)r * 128 + nt * 16 + ln] = hreg[nt][i];
    }
}

// ----------------------------------------------- wire -> face masked pooling
__global__ __launch_bounds__(256) void k_face(
    const float* __restrict__ faces, const float* __restrict__ feat_wire,
    const int* __restrict__ wire_index, const int* __restrict__ w_len,
    float* __restrict__ feat_face, _Float16* __restrict__ ff16, int n_f) {
  const int f = blockIdx.x * 8 + (threadIdx.x >> 5);
  if (f >= n_f) return;
  const int lane = threadIdx.x & 31;
  float acc[4] = {0.f, 0.f, 0.f, 0.f};
  const int L = w_len[f];
  for (int j = 0; j < 16; ++j) {
    if (j < L) {
      const int w = wire_index[f * 16 + j];
      const float* p = feat_wire + (size_t)w * 128;
#pragma unroll
      for (int c = 0; c < 4; ++c) acc[c] += p[lane + 32 * c];
    }
  }
#pragma unroll
  for (int c = 0; c < 4; ++c) {
    const int k = lane + 32 * c;
    float fv = faces[(size_t)f * 128 + k];
    feat_face[(size_t)f * 256 + k] = fv;
    ff16[(size_t)f * 256 + k] = (_Float16)fv;
    feat_face[(size_t)f * 256 + 128 + k] = acc[c];
    ff16[(size_t)f * 256 + 128 + k] = (_Float16)acc[c];
  }
}

// ------------------------------------- adj_face MLP: one wave per face (16 nbrs)
__global__ __launch_bounds__(64) void k_mlp(
    const _Float16* __restrict__ ff16, const int* __restrict__ face_index,
    const int* __restrict__ a_len, const _Float16* __restrict__ W1,
    const float* __restrict__ g1, const float* __restrict__ be1,
    const _Float16* __restrict__ W2, const float* __restrict__ g2,
    const float* __restrict__ be2, const _Float16* __restrict__ W3,
    const float* __restrict__ b3, float* __restrict__ out3, int n_f) {
  const int wv = threadIdx.x >> 5;
  const int lane = threadIdx.x & 31;
  const int ln = lane & 15, hi = lane >> 4;
  const int f = blockIdx.x * 2 + wv;
  if (f >= n_f) return;

  // warm weight caches
  prefetch_weights(W1, 512 * 256, threadIdx.x, 64);
  prefetch_weights(W2, 256 * 512, threadIdx.x, 64);
  prefetch_weights(W3, 256 * 256, threadIdx.x, 64);

  __shared__ _Float16 s_a[2][16][264];   // A0 for GEMM1, reused as h2
  __shared__ _Float16 s_h1[2][16][520];  // h1 (16x512)
  __shared__ float s_mu[2][16], s_rs[2][16];
  _Float16(*A0)[264] = s_a[wv];
  _Float16(*H1)[520] = s_h1[wv];
  float* MU = s_mu[wv];
  float* RS = s_rs[wv];

  // stage gathered rows: row ln, half hi (128 halves = 16 x v8h)
  {
    const int idx = face_index[f * 16 + ln];
    const v8h* src = (const v8h*)(ff16 + (size_t)idx * 256 + hi * 128);
    v8h* dst = (v8h*)(&A0[ln][hi * 128]);
#pragma unroll
    for (int q = 0; q < 16; ++q) dst[q] = src[q];
  }

  v16h a1[8];
#pragma unroll
  for (int kt = 0; kt < 8; ++kt) a1[kt] = load_frag(&A0[ln][kt * 32 + 8 * hi]);

  // ---- GEMM1: 16x256 @ W1^T -> 16x512, LN stats on the fly
  float s1[8], s2[8];
#pragma unroll
  for (int i = 0; i < 8; ++i) { s1[i] = 0.f; s2[i] = 0.f; }
  for (int nt = 0; nt < 32; ++nt) {
    v16h bf[8];
#pragma unroll
    for (int kt = 0; kt < 8; ++kt)
      bf[kt] = load_bfrag(W1, 256, nt * 16 + ln, kt * 32 + 8 * hi);
    pin8(bf[0], bf[1], bf[2], bf[3], bf[4], bf[5], bf[6], bf[7]);
    v8f acc;
#pragma unroll
    for (int i = 0; i < 8; ++i) acc[i] = 0.f;
#pragma unroll
    for (int kt = 0; kt < 8; ++kt) acc = wmma_f16(a1[kt], bf[kt], acc);
#pragma unroll
    for (int i = 0; i < 8; ++i) {
      float v = acc[i];
      s1[i] += v; s2[i] += v * v;
      H1[i + 8 * hi][nt * 16 + ln] = (_Float16)v;
    }
  }
#pragma unroll
  for (int i = 0; i < 8; ++i)
    for (int off = 1; off < 16; off <<= 1) {
      s1[i] += __shfl_xor(s1[i], off, 32);
      s2[i] += __shfl_xor(s2[i], off, 32);
    }
  if (ln == 0) {
#pragma unroll
    for (int i = 0; i < 8; ++i) {
      float mu = s1[i] * (1.f / 512.f);
      float var = s2[i] * (1.f / 512.f) - mu * mu;
      MU[i + 8 * hi] = mu;
      RS[i + 8 * hi] = rsqrtf(var + 1e-5f);
    }
  }
  for (int m = 0; m < 16; ++m) {
    const float mu = MU[m], rs = RS[m];
    for (int k = lane; k < 512; k += 32) {
      float x = (float)H1[m][k];
      float y = (x - mu) * rs * g1[k] + be1[k];
      H1[m][k] = (_Float16)fmaxf(y, 0.f);
    }
  }

  // ---- GEMM2: 16x512 @ W2^T -> 16x256 (K chunked by 8 tiles)
  _Float16(*H2)[264] = A0;  // A0 dead, reuse
#pragma unroll
  for (int i = 0; i < 8; ++i) { s1[i] = 0.f; s2[i] = 0.f; }
  for (int nt = 0; nt < 16; ++nt) {
    v8f acc;
#pragma unroll
    for (int i = 0; i < 8; ++i) acc[i] = 0.f;
#pragma unroll
    for (int half = 0; half < 2; ++half) {
      v16h bf[8];
#pragma unroll
      for (int k8 = 0; k8 < 8; ++k8)
        bf[k8] = load_bfrag(W2, 512, nt * 16 + ln, (half * 8 + k8) * 32 + 8 * hi);
      pin8(bf[0], bf[1], bf[2], bf[3], bf[4], bf[5], bf[6], bf[7]);
#pragma unroll
      for (int k8 = 0; k8 < 8; ++k8)
        acc = wmma_f16(load_frag(&H1[ln][(half * 8 + k8) * 32 + 8 * hi]), bf[k8],
                       acc);
    }
#pragma unroll
    for (int i = 0; i < 8; ++i) {
      float v = acc[i];
      s1[i] += v; s2[i] += v * v;
      H2[i + 8 * hi][nt * 16 + ln] = (_Float16)v;
    }
  }
#pragma unroll
  for (int i = 0; i < 8; ++i)
    for (int off = 1; off < 16; off <<= 1) {
      s1[i] += __shfl_xor(s1[i], off, 32);
      s2[i] += __shfl_xor(s2[i], off, 32);
    }
  if (ln == 0) {
#pragma unroll
    for (int i = 0; i < 8; ++i) {
      float mu = s1[i] * (1.f / 256.f);
      float var = s2[i] * (1.f / 256.f) - mu * mu;
      MU[i + 8 * hi] = mu;
      RS[i + 8 * hi] = rsqrtf(var + 1e-5f);
    }
  }
  for (int m = 0; m < 16; ++m) {
    const float mu = MU[m], rs = RS[m];
    for (int k = lane; k < 256; k += 32) {
      float x = (float)H2[m][k];
      float y = (x - mu) * rs * g2[k] + be2[k];
      H2[m][k] = (_Float16)fmaxf(y, 0.f);
    }
  }

  // ---- GEMM3: 16x256 @ W3^T + b3, mask rows >= adj_len, sum over neighbors
  const int alen = a_len[f];
  for (int nt = 0; nt < 16; ++nt) {
    v16h bf[8];
#pragma unroll
    for (int kt = 0; kt < 8; ++kt)
      bf[kt] = load_bfrag(W3, 256, nt * 16 + ln, kt * 32 + 8 * hi);
    const float bv = b3[nt * 16 + ln];
    pin8(bf[0], bf[1], bf[2], bf[3], bf[4], bf[5], bf[6], bf[7]);
    v8f acc;
#pragma unroll
    for (int i = 0; i < 8; ++i) acc[i] = bv;
#pragma unroll
    for (int kt = 0; kt < 8; ++kt)
      acc = wmma_f16(load_frag(&H2[ln][kt * 32 + 8 * hi]), bf[kt], acc);
    float p = 0.f;
#pragma unroll
    for (int i = 0; i < 8; ++i)
      if (i + 8 * hi < alen) p += acc[i];
    p += __shfl_xor(p, 16, 32);
    if (hi == 0) out3[(size_t)f * 256 + nt * 16 + ln] = p;
  }
}

// ------------------------------------------- last layer: 512 -> 128, LN, ReLU
__global__ __launch_bounds__(64) void k_out(
    const float* __restrict__ feat_face, const float* __restrict__ out3,
    const _Float16* __restrict__ W4, const float* __restrict__ g3,
    const float* __restrict__ be3, float* __restrict__ out, int n_f) {
  const int wv = threadIdx.x >> 5;
  const int lane = threadIdx.x & 31;
  const int ln = lane & 15, hi = lane >> 4;
  const int row0 = (blockIdx.x * 2 + wv) * 16;
  if (row0 >= n_f) return;

  prefetch_weights(W4, 128 * 512, threadIdx.x, 64);

  __shared__ _Float16 s_a[2][16][520];
  __shared__ float s_h[2][16][136];
  __shared__ float s_mu[2][16], s_rs[2][16];
  _Float16(*A)[520] = s_a[wv];
  float(*Hh)[136] = s_h[wv];
  float* MU = s_mu[wv];
  float* RS = s_rs[wv];

  {
    int r = row0 + ln;
    size_t fr = (size_t)(r < n_f ? r : n_f - 1);
    const float* p1 = feat_face + fr * 256;
    const float* p2 = out3 + fr * 256;
    for (int q = 0; q < 128; ++q) {
      A[ln][hi * 128 + q] = (_Float16)p1[hi * 128 + q];
      A[ln][256 + hi * 128 + q] = (_Float16)p2[hi * 128 + q];
    }
  }

  float s1[8], s2[8];
#pragma unroll
  for (int i = 0; i < 8; ++i) { s1[i] = 0.f; s2[i] = 0.f; }
  for (int nt = 0; nt < 8; ++nt) {
    v8f acc;
#pragma unroll
    for (int i = 0; i < 8; ++i) acc[i] = 0.f;
#pragma unroll
    for (int half = 0; half < 2; ++half) {
      v16h bf[8];
#pragma unroll
      for (int k8 = 0; k8 < 8; ++k8)
        bf[k8] = load_bfrag(W4, 512, nt * 16 + ln, (half * 8 + k8) * 32 + 8 * hi);
      pin8(bf[0], bf[1], bf[2], bf[3], bf[4], bf[5], bf[6], bf[7]);
#pragma unroll
      for (int k8 = 0; k8 < 8; ++k8)
        acc = wmma_f16(load_frag(&A[ln][(half * 8 + k8) * 32 + 8 * hi]), bf[k8],
                       acc);
    }
#pragma unroll
    for (int i = 0; i < 8; ++i) {
      float v = acc[i];
      s1[i] += v; s2[i] += v * v;
      Hh[i + 8 * hi][nt * 16 + ln] = v;
    }
  }
#pragma unroll
  for (int i = 0; i < 8; ++i)
    for (int off = 1; off < 16; off <<= 1) {
      s1[i] += __shfl_xor(s1[i], off, 32);
      s2[i] += __shfl_xor(s2[i], off, 32);
    }
  if (ln == 0) {
#pragma unroll
    for (int i = 0; i < 8; ++i) {
      float mu = s1[i] * (1.f / 128.f);
      float var = s2[i] * (1.f / 128.f) - mu * mu;
      MU[i + 8 * hi] = mu;
      RS[i + 8 * hi] = rsqrtf(var + 1e-5f);
    }
  }
  for (int m = 0; m < 16; ++m) {
    int r = row0 + m;
    if (r >= n_f) break;
    const float mu = MU[m], rs = RS[m];
    for (int k = lane; k < 128; k += 32) {
      float y = (Hh[m][k] - mu) * rs * g3[k] + be3[k];
      out[(size_t)r * 128 + k] = fmaxf(y, 0.f);
    }
  }
}

extern "C" void kernel_launch(void* const* d_in, const int* in_sizes, int n_in,
                              void* d_out, int out_size, void* d_ws, size_t ws_size,
                              hipStream_t stream) {
  const float* edges = (const float*)d_in[0];
  const float* faces = (const float*)d_in[1];
  const int* edge_index = (const int*)d_in[2];
  const int* wire_index = (const int*)d_in[3];
  const int* face_index = (const int*)d_in[4];
  const int* e_len = (const int*)d_in[5];
  const int* w_len = (const int*)d_in[6];
  const int* a_len = (const int*)d_in[7];
  const float* h0 = (const float*)d_in[8];
  const float* W_ih = (const float*)d_in[9];
  const float* W_hh = (const float*)d_in[10];
  const float* b_ih = (const float*)d_in[11];
  const float* b_hh = (const float*)d_in[12];
  const float* W1 = (const float*)d_in[13];
  const float* g1 = (const float*)d_in[14];
  const float* be1 = (const float*)d_in[15];
  const float* W2 = (const float*)d_in[16];
  const float* g2 = (const float*)d_in[17];
  const float* be2 = (const float*)d_in[18];
  const float* W3 = (const float*)d_in[19];
  const float* b3 = (const float*)d_in[20];
  const float* W4 = (const float*)d_in[21];
  const float* g3 = (const float*)d_in[22];
  const float* be3 = (const float*)d_in[23];

  const int n_w = in_sizes[5];
  const int n_f = in_sizes[6];

  char* ws = (char*)d_ws;
  size_t off = 0;
  auto take = [&](size_t bytes) {
    char* p = ws + off;
    off = (off + bytes + 255) & ~(size_t)255;
    return p;
  };
  float* feat_wire = (float*)take((size_t)n_w * 128 * 4);
  float* feat_face = (float*)take((size_t)n_f * 256 * 4);
  _Float16* ff16 = (_Float16*)take((size_t)n_f * 256 * 2);
  float* out3 = (float*)take((size_t)n_f * 256 * 4);
  _Float16* Wih16 = (_Float16*)take(128 * 128 * 2);
  _Float16* Whh16 = (_Float16*)take(128 * 128 * 2);
  _Float16* W1h = (_Float16*)take(512 * 256 * 2);
  _Float16* W2h = (_Float16*)take(256 * 512 * 2);
  _Float16* W3h = (_Float16*)take(256 * 256 * 2);
  _Float16* W4h = (_Float16*)take(128 * 512 * 2);

  auto cv = [&](const float* s, _Float16* d, int n) {
    k_cvt<<<(n + 255) / 256, 256, 0, stream>>>(s, d, n);
  };
  cv(W_ih, Wih16, 128 * 128);
  cv(W_hh, Whh16, 128 * 128);
  cv(W1, W1h, 512 * 256);
  cv(W2, W2h, 256 * 512);
  cv(W3, W3h, 256 * 256);
  cv(W4, W4h, 128 * 512);

  {
    int tiles = (n_w + 15) / 16;
    int blocks = (tiles + 3) / 4;
    k_rnn<<<blocks, 128, 0, stream>>>(edges, edge_index, e_len, h0, Wih16, Whh16,
                                      b_ih, b_hh, feat_wire, n_w);
  }
  k_face<<<(n_f + 7) / 8, 256, 0, stream>>>(faces, feat_wire, wire_index, w_len,
                                            feat_face, ff16, n_f);
  k_mlp<<<(n_f + 1) / 2, 64, 0, stream>>>(ff16, face_index, a_len, W1h, g1, be1,
                                          W2h, g2, be2, W3h, b3, out3, n_f);
  {
    int tiles = (n_f + 15) / 16;
    int blocks = (tiles + 1) / 2;
    k_out<<<blocks, 64, 0, stream>>>(feat_face, out3, W4h, g3, be3, (float*)d_out,
                                     n_f);
  }
}
